// FlashAttention_74088185856446
// MI455X (gfx1250) — matrix operand order repaired and checked by reading, not compiled, not run on hardware
//
#include <hip/hip_runtime.h>
#include <math.h>

typedef __attribute__((ext_vector_type(16))) _Float16 v16h;
typedef __attribute__((ext_vector_type(4)))  _Float16 v4h;
typedef __attribute__((ext_vector_type(8)))  float    v8f;

#define HEAD  64
#define TLEN  2048
#define NHEAD 16
#define BATCH 2
#define KBLK  32
#define QTILE 32   // queries per wave (two 16-row WMMA sub-tiles)
#define WPB   8    // waves per block

// ---------------------------------------------------------------------------
// 16-lane max-reduction butterfly via DPP (pure VALU, no LDS traffic).
// ---------------------------------------------------------------------------
__device__ __forceinline__ float dpp_max_bcast16(float v) {
    int x;
    x = __builtin_amdgcn_update_dpp(0, __float_as_int(v), 0xB1, 0xF, 0xF, true);
    v = fmaxf(v, __int_as_float(x));
    x = __builtin_amdgcn_update_dpp(0, __float_as_int(v), 0x4E, 0xF, 0xF, true);
    v = fmaxf(v, __int_as_float(x));
    x = __builtin_amdgcn_update_dpp(0, __float_as_int(v), 0x141, 0xF, 0xF, true);
    v = fmaxf(v, __int_as_float(x));
    x = __builtin_amdgcn_update_dpp(0, __float_as_int(v), 0x140, 0xF, 0xF, true);
    v = fmaxf(v, __int_as_float(x));
    return v;
}

// ---------------------------------------------------------------------------
// Pre-pass 1: fp32 -> f16 elementwise (optionally scaled), vectorized x4.
// ---------------------------------------------------------------------------
__global__ __launch_bounds__(256) void
cvt_f16_kernel(const float* __restrict__ src, _Float16* __restrict__ dst,
               float scale, int n4)
{
    int i = blockIdx.x * 256 + threadIdx.x;
    if (i < n4) {
        float4 f = ((const float4*)src)[i];
        v4h h;
        h[0] = (_Float16)(f.x * scale);
        h[1] = (_Float16)(f.y * scale);
        h[2] = (_Float16)(f.z * scale);
        h[3] = (_Float16)(f.w * scale);
        ((v4h*)dst)[i] = h;
    }
}

// ---------------------------------------------------------------------------
// Pre-pass 2: V (bh, T, 64) fp32  ->  Vt (bh, 64, T) f16, via LDS tile.
// ---------------------------------------------------------------------------
__global__ __launch_bounds__(256) void
transpose_v_kernel(const float* __restrict__ V, _Float16* __restrict__ Vt)
{
    __shared__ _Float16 Lt[64][72];
    const int bh = blockIdx.x / (TLEN / 64);
    const int t0 = (blockIdx.x % (TLEN / 64)) * 64;

    const float* src = V + ((size_t)bh * TLEN + t0) * HEAD;
#pragma unroll
    for (int i = 0; i < 16; ++i) {
        int e = threadIdx.x + i * 256;
        int t = e >> 6, d = e & 63;
        Lt[d][t] = (_Float16)src[e];
    }
    __syncthreads();
    _Float16* dst = Vt + (size_t)bh * HEAD * TLEN + t0;
#pragma unroll
    for (int i = 0; i < 16; ++i) {
        int e = threadIdx.x + i * 256;
        int d = e >> 6, t = e & 63;
        dst[(size_t)d * TLEN + t] = Lt[d][t];
    }
}

// ---------------------------------------------------------------------------
// Main kernel: one wave32 owns a 32-query tile (two 16-row WMMA sub-tiles
// sharing the same K and V fragments), streaming keys in blocks of 32.
// Per key-block: 18 WMMAs (8 QK^T + 2 row-sum-vs-ones + 8 PV) against
// 16 b128 K/V loads. Mask folded into WMMA C init; row-max via DPP.
// ---------------------------------------------------------------------------
__global__ __launch_bounds__(256) void
FlashAttention_74088185856446_kernel(const _Float16* __restrict__ Qh,  // pre-scaled
                                     const _Float16* __restrict__ Kh,
                                     const _Float16* __restrict__ Vt,  // (bh, d, T)
                                     const float*    __restrict__ mask,
                                     float*          __restrict__ Out)
{
    const int lane = threadIdx.x & 31;
    const int wave = threadIdx.x >> 5;
    const int tile = blockIdx.x * WPB + wave;          // 32-query tile id
    const int tiles_per_head = TLEN / QTILE;
    const int bh = tile / tiles_per_head;
    const int q0 = (tile % tiles_per_head) * QTILE;
    const int b  = bh / NHEAD;

    const int n = lane & 15;
    const int g = lane >> 4;

    __shared__ _Float16 Pt[WPB][2][16][KBLK];          // per-wave P staging

    // ---- Q tiles as A-fragments (one contiguous 32B run per lane each)
    v16h a0[2], a1[2];
#pragma unroll
    for (int t = 0; t < 2; ++t) {
        const _Float16* qrow = Qh + ((size_t)bh * TLEN + q0 + 16*t + n) * HEAD;
#pragma unroll
        for (int h = 0; h < 8; ++h) {
            a0[t][h]     = qrow[     16*g + h];
            a0[t][h + 8] = qrow[ 8 + 16*g + h];
            a1[t][h]     = qrow[32 + 16*g + h];
            a1[t][h + 8] = qrow[40 + 16*g + h];
        }
    }

    v16h onesB;
#pragma unroll
    for (int h = 0; h < 16; ++h) onesB[h] = (_Float16)1.0f;

    v8f o[2][4], l_acc[2];
    float m_run[2][8];
#pragma unroll
    for (int t = 0; t < 2; ++t) {
        l_acc[t] = (v8f){0.f,0.f,0.f,0.f,0.f,0.f,0.f,0.f};
#pragma unroll
        for (int j = 0; j < 4; ++j)
            o[t][j] = (v8f){0.f,0.f,0.f,0.f,0.f,0.f,0.f,0.f};
#pragma unroll
        for (int i = 0; i < 8; ++i) m_run[t][i] = -1e30f;
    }

    for (int kb = 0; kb < TLEN; kb += KBLK) {
        // ---- K B-fragments, shared by both query sub-tiles (2x b128 each)
        const _Float16* krow_lo = Kh + ((size_t)bh * TLEN + kb + n) * HEAD + 16*g;
        const _Float16* krow_hi = krow_lo + (size_t)16 * HEAD;
        v16h bl0 = *(const v16h*)(krow_lo);
        v16h bl1 = *(const v16h*)(krow_lo + 32);
        v16h bh0 = *(const v16h*)(krow_hi);
        v16h bh1 = *(const v16h*)(krow_hi + 32);

        const float mk_lo = mask[(size_t)b * TLEN + kb + n];
        const float mk_hi = mask[(size_t)b * TLEN + kb + 16 + n];

#pragma unroll
        for (int t = 0; t < 2; ++t) {
            // ---- S = Q*K^T (mask as C init; Q carries 1/sqrt(hs))
            v8f s_lo, s_hi;
#pragma unroll
            for (int i = 0; i < 8; ++i) { s_lo[i] = mk_lo; s_hi[i] = mk_hi; }
            s_lo = __builtin_amdgcn_wmma_f32_16x16x32_f16(false, a0[t], false, bl0,
                                                          (short)0, s_lo, false, false);
            s_lo = __builtin_amdgcn_wmma_f32_16x16x32_f16(false, a1[t], false, bl1,
                                                          (short)0, s_lo, false, false);
            s_hi = __builtin_amdgcn_wmma_f32_16x16x32_f16(false, a0[t], false, bh0,
                                                          (short)0, s_hi, false, false);
            s_hi = __builtin_amdgcn_wmma_f32_16x16x32_f16(false, a1[t], false, bh1,
                                                          (short)0, s_hi, false, false);

            // ---- online softmax: DPP row-max, exp, write P to LDS
            float alpha[8];
#pragma unroll
            for (int i = 0; i < 8; ++i) {
                float v = dpp_max_bcast16(fmaxf(s_lo[i], s_hi[i]));
                float m_new = fmaxf(m_run[t][i], v);
                alpha[i]    = __expf(m_run[t][i] - m_new);
                m_run[t][i] = m_new;
                float pl = __expf(s_lo[i] - m_new);
                float ph = __expf(s_hi[i] - m_new);
                Pt[wave][t][i + 8*g][n]      = (_Float16)pl;
                Pt[wave][t][i + 8*g][n + 16] = (_Float16)ph;
            }
#pragma unroll
            for (int i = 0; i < 8; ++i) {
                o[t][0][i] *= alpha[i];
                o[t][1][i] *= alpha[i];
                o[t][2][i] *= alpha[i];
                o[t][3][i] *= alpha[i];
                l_acc[t][i] *= alpha[i];
            }
        }

        // ---- reload P tiles as A-fragments; row-sum WMMAs
        v16h pA[2];
#pragma unroll
        for (int t = 0; t < 2; ++t) {
#pragma unroll
            for (int h = 0; h < 8; ++h) {
                pA[t][h]     = Pt[wave][t][n][    16*g + h];
                pA[t][h + 8] = Pt[wave][t][n][8 + 16*g + h];
            }
            l_acc[t] = __builtin_amdgcn_wmma_f32_16x16x32_f16(false, pA[t], false, onesB,
                                                              (short)0, l_acc[t], false, false);
        }

        // ---- O += P * V, one V fragment reused by both sub-tiles
        const _Float16* vt = Vt + ((size_t)bh * HEAD + n) * TLEN + kb + 16*g;
#pragma unroll
        for (int j = 0; j < 4; ++j) {
            v16h vB = *(const v16h*)(vt + (size_t)(16*j) * TLEN);
            o[0][j] = __builtin_amdgcn_wmma_f32_16x16x32_f16(false, pA[0], false, vB,
                                                             (short)0, o[0][j], false, false);
            o[1][j] = __builtin_amdgcn_wmma_f32_16x16x32_f16(false, pA[1], false, vB,
                                                             (short)0, o[1][j], false, false);
        }
    }

    // ---- normalize and store (coalesced fp32)
#pragma unroll
    for (int t = 0; t < 2; ++t) {
#pragma unroll
        for (int i = 0; i < 8; ++i) {
            float inv = 1.0f / l_acc[t][i];
            const size_t row = ((size_t)bh * TLEN + q0 + 16*t + i + 8*g) * HEAD;
            Out[row +      n] = o[t][0][i] * inv;
            Out[row + 16 + n] = o[t][1][i] * inv;
            Out[row + 32 + n] = o[t][2][i] * inv;
            Out[row + 48 + n] = o[t][3][i] * inv;
        }
    }
}

extern "C" void kernel_launch(void* const* d_in, const int* in_sizes, int n_in,
                              void* d_out, int out_size, void* d_ws, size_t ws_size,
                              hipStream_t stream) {
    const float* Q    = (const float*)d_in[0];
    const float* K    = (const float*)d_in[1];
    const float* V    = (const float*)d_in[2];
    const float* mask = (const float*)d_in[3];
    float* Out        = (float*)d_out;

    const size_t N = (size_t)BATCH * NHEAD * TLEN * HEAD;    // 4,194,304 elems
    _Float16* Qh = (_Float16*)d_ws;                          //  8 MB
    _Float16* Kh = Qh + N;                                   //  8 MB
    _Float16* Vt = Kh + N;                                   //  8 MB (transposed)

    const int n4       = (int)(N / 4);
    const int cvt_blks = (n4 + 255) / 256;
    const float scale  = 0.125f;                             // 1/sqrt(64)

    cvt_f16_kernel<<<cvt_blks, 256, 0, stream>>>(Q, Qh, scale, n4);
    cvt_f16_kernel<<<cvt_blks, 256, 0, stream>>>(K, Kh, 1.0f, n4);
    transpose_v_kernel<<<BATCH * NHEAD * (TLEN / 64), 256, 0, stream>>>(V, Vt);

    const int tiles  = BATCH * NHEAD * (TLEN / QTILE);       // 2048 query tiles
    const int blocks = tiles / WPB;                          // 256 blocks
    FlashAttention_74088185856446_kernel<<<blocks, 256, 0, stream>>>(Qh, Kh, Vt, mask, Out);
}
